// BrainGAT_58402965291709
// MI455X (gfx1250) — matrix operand order repaired and checked by reading, not compiled, not run on hardware
//
#include <hip/hip_runtime.h>
#include <hip/hip_bf16.h>
#include <math.h>

typedef __attribute__((ext_vector_type(16))) __bf16 v16bf;
typedef __attribute__((ext_vector_type(8)))  float  v8f;

// Problem constants (match reference)
constexpr int N_NODES  = 32768;
constexpr int E_EDGES  = 524288;
constexpr int G_GRAPHS = 64;
constexpr int IN_F     = 4;
constexpr int HEADS    = 8;
constexpr int CH       = 64;
constexpr int HC       = 512;   // HEADS*CH
constexpr int OUT_F    = 384;
constexpr float NEG_SLOPE = 0.2f;
constexpr int E_TOT    = E_EDGES + N_NODES;  // with self-loops

// ---------- helpers ----------
__device__ __forceinline__ float lrelu(float v) { return v > 0.f ? v : NEG_SLOPE * v; }
// monotone float <-> uint mapping for atomicMax on floats (0u decodes below any real value)
__device__ __forceinline__ unsigned fenc(float f) {
  unsigned u = __float_as_uint(f);
  return (u & 0x80000000u) ? ~u : (u | 0x80000000u);
}
__device__ __forceinline__ float fdec(unsigned u) {
  return (u & 0x80000000u) ? __uint_as_float(u & 0x7FFFFFFFu) : __uint_as_float(~u);
}

// ---------- conversion kernels ----------
// x [n,k_in] f32 -> hbf [n,k_pad] bf16 (zero padded)
__global__ void k_convert_pad(const float* __restrict__ x, __bf16* __restrict__ hbf,
                              int n, int k_in, int k_pad) {
  int t = blockIdx.x * blockDim.x + threadIdx.x;
  if (t >= n * k_pad) return;
  int row = t / k_pad, k = t - row * k_pad;
  float v = (k < k_in) ? x[row * k_in + k] : 0.f;
  hbf[t] = (__bf16)v;
}

// W [k_in, m] f32 (row-major) -> Wt [m, k_pad] bf16 (transposed, zero padded)
__global__ void k_w_transpose(const float* __restrict__ W, __bf16* __restrict__ wt,
                              int k_in, int m, int k_pad) {
  int t = blockIdx.x * blockDim.x + threadIdx.x;
  if (t >= m * k_pad) return;
  int row = t / k_pad, k = t - row * k_pad;   // row = output feature, k = input feature
  float v = (k < k_in) ? W[k * m + row] : 0.f;
  wt[t] = (__bf16)v;
}

// ---------- WMMA GEMM: C[n,m] = A[n,kpad](bf16) * Bt[m,kpad]^T (bf16), f32 accumulate.
// One wave computes a 16(M) x 64(N) strip: A fragment loaded once per K-step and
// reused by 4 WMMAs (2.5 b128 loads per WMMA instead of 4). Requires m % 64 == 0.
__global__ __launch_bounds__(256)
void k_gemm_wmma(const __bf16* __restrict__ A, const __bf16* __restrict__ Bt,
                 float* __restrict__ Cout, int n, int kpad, int m) {
  const int lane    = threadIdx.x & 31;
  const int wid     = blockIdx.x * (blockDim.x >> 5) + (threadIdx.x >> 5);
  const int nstrips = m >> 6;                      // 64-wide output strips
  if (wid >= (n >> 4) * nstrips) return;
  const int mt  = wid / nstrips;
  const int ns  = wid - mt * nstrips;

  const int arow = mt * 16 + (lane & 15);
  const int ka   = (lane >> 4) * 8;    // A: lanes 16-31 hold the K=16..31 half (base = 2*ka)
  const int kb   = (lane >> 4) * 16;   // B: lanes 16-31 hold the K=16..31 half

  const unsigned* __restrict__ Arow = (const unsigned*)(A + (size_t)arow * kpad);
  const unsigned* __restrict__ Bcol[4];
#pragma unroll
  for (int t = 0; t < 4; ++t) {
    int bcol = ns * 64 + t * 16 + (lane & 15);
    Bcol[t] = (const unsigned*)(Bt + (size_t)bcol * kpad);
  }

  v8f acc0 = {}, acc1 = {}, acc2 = {}, acc3 = {};
  for (int k0 = 0; k0 < kpad; k0 += 32) {
    union { v16bf v; unsigned u[8]; } a;
#pragma unroll
    for (int r = 0; r < 4; ++r) {
      a.u[r]     = Arow[(k0 + 2 * ka + 2 * r) >> 1];        // K = base..base+7
      a.u[r + 4] = Arow[(k0 + 2 * ka + 8 + 2 * r) >> 1];    // K = base+8..base+15
    }
    union { v16bf v; unsigned u[8]; } b0, b1, b2, b3;
#pragma unroll
    for (int r = 0; r < 8; ++r) {
      int ki = (k0 + kb + 2 * r) >> 1;
      b0.u[r] = Bcol[0][ki];
      b1.u[r] = Bcol[1][ki];
      b2.u[r] = Bcol[2][ki];
      b3.u[r] = Bcol[3][ki];
    }
    acc0 = __builtin_amdgcn_wmma_f32_16x16x32_bf16(false, a.v, false, b0.v, (short)0, acc0, false, false);
    acc1 = __builtin_amdgcn_wmma_f32_16x16x32_bf16(false, a.v, false, b1.v, (short)0, acc1, false, false);
    acc2 = __builtin_amdgcn_wmma_f32_16x16x32_bf16(false, a.v, false, b2.v, (short)0, acc2, false, false);
    acc3 = __builtin_amdgcn_wmma_f32_16x16x32_bf16(false, a.v, false, b3.v, (short)0, acc3, false, false);
  }

  // C/D layout: VGPR r -> lanes 0-15: (M=r, N=lane); lanes 16-31: (M=r+8, N=lane-16)
  const int rbase = mt * 16 + (lane >> 4) * 8;
  const int cbase = ns * 64 + (lane & 15);
#pragma unroll
  for (int r = 0; r < 8; ++r) {
    float* crow = Cout + (size_t)(rbase + r) * m + cbase;
    crow[0]  = acc0[r];
    crow[16] = acc1[r];
    crow[32] = acc2[r];
    crow[48] = acc3[r];
  }
}

// ---------- per-node attention logits: als/ald [n,heads] ----------
__global__ void k_attn_logits(const float* __restrict__ xw,
                              const float* __restrict__ a_s, const float* __restrict__ a_d,
                              float* __restrict__ als, float* __restrict__ ald,
                              int n, int heads, int ch) {
  int t = blockIdx.x * blockDim.x + threadIdx.x;
  if (t >= n * heads) return;
  int node = t / heads, h = t - node * heads;
  const float* row = xw + (size_t)node * heads * ch + (size_t)h * ch;
  const float* asr = a_s + h * ch;
  const float* adr = a_d + h * ch;
  float s = 0.f, d = 0.f;
  for (int c = 0; c < ch; ++c) { float v = row[c]; s += v * asr[c]; d += v * adr[c]; }
  als[t] = s; ald[t] = d;
}

// ---------- edge pass 1: segment max ----------
__global__ void k_edge_max(const int* __restrict__ src_e, const int* __restrict__ dst_e,
                           const float* __restrict__ als, const float* __restrict__ ald,
                           unsigned* __restrict__ maxe, int e, int n, int heads) {
  int t = blockIdx.x * blockDim.x + threadIdx.x;
  if (t >= e + n) return;
  int s = (t < e) ? src_e[t] : (t - e);
  int d = (t < e) ? dst_e[t] : (t - e);
  for (int h = 0; h < heads; ++h) {
    float ev = lrelu(als[s * heads + h] + ald[d * heads + h]);
    atomicMax(&maxe[d * heads + h], fenc(ev));
  }
}

// ---------- edge pass 2: segment sum of exp ----------
__global__ void k_edge_den(const int* __restrict__ src_e, const int* __restrict__ dst_e,
                           const float* __restrict__ als, const float* __restrict__ ald,
                           const unsigned* __restrict__ maxe, float* __restrict__ den,
                           int e, int n, int heads) {
  int t = blockIdx.x * blockDim.x + threadIdx.x;
  if (t >= e + n) return;
  int s = (t < e) ? src_e[t] : (t - e);
  int d = (t < e) ? dst_e[t] : (t - e);
  for (int h = 0; h < heads; ++h) {
    float ev = lrelu(als[s * heads + h] + ald[d * heads + h]);
    float p  = __expf(ev - fdec(maxe[d * heads + h]));
    atomicAdd(&den[d * heads + h], p);
  }
}

// ---------- edge pass 3: weighted scatter (one wave per edge) ----------
__global__ __launch_bounds__(256)
void k_edge_agg(const int* __restrict__ src_e, const int* __restrict__ dst_e,
                const float* __restrict__ als, const float* __restrict__ ald,
                const unsigned* __restrict__ maxe, const float* __restrict__ den,
                const float* __restrict__ xw, float* __restrict__ acc,
                int e, int n, int heads, int hc) {
  const int lane = threadIdx.x & 31;
  const int eid  = blockIdx.x * (blockDim.x >> 5) + (threadIdx.x >> 5);
  if (eid >= e + n) return;
  const int s = (eid < e) ? src_e[eid] : (eid - e);
  const int d = (eid < e) ? dst_e[eid] : (eid - e);
  const int chunk = hc >> 5;          // features per lane (16 for HC=512, 2 for HC=64)
  const int fbase = lane * chunk;
  const int ch    = hc / heads;
  const int h     = fbase / ch;       // constant per lane
  float ev    = lrelu(als[s * heads + h] + ald[d * heads + h]);
  float alpha = __expf(ev - fdec(maxe[d * heads + h])) / den[d * heads + h];
  const float* xr = xw  + (size_t)s * hc;
  float* orow     = acc + (size_t)d * hc;
  for (int j = 0; j < chunk; ++j) {
    int f = fbase + j;
    atomicAdd(&orow[f], alpha * xr[f]);
  }
}

// ---------- bias + ELU; write bf16 (next layer input) and/or f32 ----------
__global__ void k_bias_elu(const float* __restrict__ acc, const float* __restrict__ bias,
                           __bf16* __restrict__ out_bf, float* __restrict__ out_f32,
                           int n, int hc) {
  int t = blockIdx.x * blockDim.x + threadIdx.x;
  if (t >= n * hc) return;
  int f = t % hc;
  float v = acc[t] + bias[f];
  float r = v > 0.f ? v : expm1f(v);
  if (out_bf)  out_bf[t]  = (__bf16)r;
  if (out_f32) out_f32[t] = r;
}

// ---------- global mean pool: graphs are contiguous 512-node blocks ----------
__global__ void k_pool(const float* __restrict__ h3, float* __restrict__ pooled) {
  __shared__ float sm[256];
  int g = blockIdx.x;
  int c = threadIdx.x & 63;
  int part = threadIdx.x >> 6;
  float s = 0.f;
  for (int nloc = part; nloc < 512; nloc += 4)
    s += h3[(size_t)(g * 512 + nloc) * 64 + c];
  sm[threadIdx.x] = s;
  __syncthreads();
  if (part == 0)
    pooled[g * 64 + c] = (sm[c] + sm[64 + c] + sm[128 + c] + sm[192 + c]) * (1.0f / 512.0f);
}

// ---------- final projection [G,64] @ [64,OUT] + bp ----------
__global__ void k_final(const float* __restrict__ pooled, const float* __restrict__ Wp,
                        const float* __restrict__ bp, float* __restrict__ out) {
  int t = blockIdx.x * blockDim.x + threadIdx.x;
  if (t >= G_GRAPHS * OUT_F) return;
  int g = t / OUT_F, o = t - g * OUT_F;
  float s = bp[o];
  for (int c = 0; c < CH; ++c) s += pooled[g * CH + c] * Wp[c * OUT_F + o];
  out[t] = s;
}

// ---------- host orchestration ----------
static inline int cdiv(int a, int b) { return (a + b - 1) / b; }

extern "C" void kernel_launch(void* const* d_in, const int* in_sizes, int n_in,
                              void* d_out, int out_size, void* d_ws, size_t ws_size,
                              hipStream_t stream) {
  (void)in_sizes; (void)n_in; (void)out_size; (void)ws_size;
  const float* x     = (const float*)d_in[0];
  const int*   ei    = (const int*)d_in[1];
  const int*   src_e = ei;
  const int*   dst_e = ei + E_EDGES;
  const float* W1  = (const float*)d_in[4];
  const float* a1s = (const float*)d_in[5];
  const float* a1d = (const float*)d_in[6];
  const float* b1  = (const float*)d_in[7];
  const float* W2  = (const float*)d_in[8];
  const float* a2s = (const float*)d_in[9];
  const float* a2d = (const float*)d_in[10];
  const float* b2  = (const float*)d_in[11];
  const float* W3  = (const float*)d_in[12];
  const float* a3s = (const float*)d_in[13];
  const float* a3d = (const float*)d_in[14];
  const float* b3  = (const float*)d_in[15];
  const float* Wp  = (const float*)d_in[16];
  const float* bp  = (const float*)d_in[17];
  float* out = (float*)d_out;

  // workspace carve-up
  char* ws = (char*)d_ws;
  size_t off = 0;
  auto carve = [&](size_t bytes) { void* p = ws + off; off = (off + bytes + 255) & ~(size_t)255; return p; };
  float*    XW   = (float*)   carve((size_t)N_NODES * HC * 4);   // gemm out / layer-3 h3 reuse
  float*    ACC  = (float*)   carve((size_t)N_NODES * HC * 4);   // aggregation accumulator
  __bf16*   HBF  = (__bf16*)  carve((size_t)N_NODES * HC * 2);   // bf16 layer input
  __bf16*   WT   = (__bf16*)  carve((size_t)HC * HC * 2);        // transposed bf16 weights
  float*    ALS  = (float*)   carve((size_t)N_NODES * HEADS * 4);
  float*    ALD  = (float*)   carve((size_t)N_NODES * HEADS * 4);
  unsigned* MAXE = (unsigned*)carve((size_t)N_NODES * HEADS * 4);
  float*    DEN  = (float*)   carve((size_t)N_NODES * HEADS * 4);
  float*    POOL = (float*)   carve((size_t)G_GRAPHS * CH * 4);

  const int B = 256;

  auto run_layer = [&](const float* Wl, int k_in, int k_pad, int m_out,
                       const float* as, const float* ad, const float* bias,
                       int heads, int ch, bool first, bool last) {
    int hc = heads * ch;
    if (first)
      k_convert_pad<<<cdiv(N_NODES * k_pad, B), B, 0, stream>>>(x, HBF, N_NODES, k_in, k_pad);
    k_w_transpose<<<cdiv(m_out * k_pad, B), B, 0, stream>>>(Wl, WT, k_in, m_out, k_pad);
    int waves = (N_NODES >> 4) * (m_out >> 6);   // 16x64 strip per wave
    k_gemm_wmma<<<cdiv(waves, 8), B, 0, stream>>>(HBF, WT, XW, N_NODES, k_pad, m_out);
    k_attn_logits<<<cdiv(N_NODES * heads, B), B, 0, stream>>>(XW, as, ad, ALS, ALD, N_NODES, heads, ch);
    hipMemsetAsync(MAXE, 0, (size_t)N_NODES * heads * 4, stream);  // fenc-domain -inf
    hipMemsetAsync(DEN,  0, (size_t)N_NODES * heads * 4, stream);
    hipMemsetAsync(ACC,  0, (size_t)N_NODES * hc * 4, stream);
    k_edge_max<<<cdiv(E_TOT, B), B, 0, stream>>>(src_e, dst_e, ALS, ALD, MAXE, E_EDGES, N_NODES, heads);
    k_edge_den<<<cdiv(E_TOT, B), B, 0, stream>>>(src_e, dst_e, ALS, ALD, MAXE, DEN, E_EDGES, N_NODES, heads);
    k_edge_agg<<<cdiv(E_TOT, 8), B, 0, stream>>>(src_e, dst_e, ALS, ALD, MAXE, DEN, XW, ACC,
                                                 E_EDGES, N_NODES, heads, hc);
    if (!last)
      k_bias_elu<<<cdiv(N_NODES * hc, B), B, 0, stream>>>(ACC, bias, HBF, (float*)nullptr, N_NODES, hc);
    else
      k_bias_elu<<<cdiv(N_NODES * hc, B), B, 0, stream>>>(ACC, bias, (__bf16*)nullptr, XW, N_NODES, hc);
  };

  // Layer 1: [N,4]->[N,512], K padded 4->32
  run_layer(W1, IN_F, 32, HC, a1s, a1d, b1, HEADS, CH, /*first=*/true,  /*last=*/false);
  // Layer 2: [N,512]->[N,512]
  run_layer(W2, HC,  HC, HC, a2s, a2d, b2, HEADS, CH, /*first=*/false, /*last=*/false);
  // Layer 3: [N,512]->[N,64], single head; h3 (f32) written into XW
  run_layer(W3, HC,  HC, CH, a3s, a3d, b3, 1, CH,     /*first=*/false, /*last=*/true);

  k_pool<<<G_GRAPHS, B, 0, stream>>>(XW, POOL);
  k_final<<<cdiv(G_GRAPHS * OUT_F, B), B, 0, stream>>>(POOL, Wp, bp, out);
}
